// RationaleExtractor_40810779247513
// MI455X (gfx1250) — matrix operand order repaired and checked by reading, not compiled, not run on hardware
//
#include <hip/hip_runtime.h>
#include <hip/hip_bf16.h>

typedef __attribute__((ext_vector_type(16))) __bf16 v16bf;
typedef __attribute__((ext_vector_type(8)))  __bf16 v8bf;
typedef __attribute__((ext_vector_type(8)))  float  v8f;

union ABFrag { v16bf v; v8bf h[2]; };

__device__ __forceinline__ v8f zero_v8f() {
  v8f z;
#pragma unroll
  for (int i = 0; i < 8; ++i) z[i] = 0.0f;
  return z;
}

// A fragment (16x32 bf16, interleaved per ISA table):
// lanes 0-15: row = lane, V0-3 = K[kb..kb+7], V4-7 = K[kb+16..kb+23]
// lanes 16-31: row = lane-16, V0-3 = K[kb+8..+15], V4-7 = K[kb+24..+31]
__device__ __forceinline__ v16bf ldA(const __bf16* base, long rstride, int row0, int kbase) {
  int lane = threadIdx.x & 31;
  const __bf16* rp = base + (long)(row0 + (lane & 15)) * rstride;
  int sel = lane >> 4;
  ABFrag u;
  u.h[0] = *(const v8bf*)(rp + kbase + sel * 8);
  u.h[1] = *(const v8bf*)(rp + kbase + 16 + sel * 8);
  return u.v;
}

// B fragment (32x16 bf16): lanes 0-15 hold K[kb..kb+7], K[kb+16..kb+23] of column (col0+lane),
// lanes 16-31 hold K[kb+8..kb+15], K[kb+24..kb+31]. Source is "transposed" storage: row-major N x K.
__device__ __forceinline__ v16bf ldBT(const __bf16* base, long rstride, int col0, int kbase) {
  int lane = threadIdx.x & 31;
  const __bf16* rp = base + (long)(col0 + (lane & 15)) * rstride;
  int sel = lane >> 4;
  ABFrag u;
  u.h[0] = *(const v8bf*)(rp + kbase + sel * 8);
  u.h[1] = *(const v8bf*)(rp + kbase + 16 + sel * 8);
  return u.v;
}

__device__ __forceinline__ v8f wmma_bf16(v16bf a, v16bf b, v8f c) {
  return __builtin_amdgcn_wmma_f32_16x16x32_bf16(false, a, false, b, (short)0, c, false, false);
}

// ---------------------------------------------------------------- prep kernels
__global__ void __launch_bounds__(256) convert_hs_kernel(const float* hs, __bf16* hsb, __bf16* hsT) {
  const int S = 2048, H = 1024;
  long i = (long)blockIdx.x * blockDim.x + threadIdx.x;
  if (i >= (long)16 * S * H) return;
  float v = hs[i];
  hsb[i] = (__bf16)v;
  long h = i % H;
  long bs = i / H;
  long s = bs % S;
  long b = bs / S;
  hsT[(b * H + h) * S + s] = (__bf16)v;
}

__global__ void __launch_bounds__(256) transposeW_kernel(const float* Win, __bf16* WT, int K, int N) {
  long i = (long)blockIdx.x * blockDim.x + threadIdx.x;
  if (i >= (long)K * N) return;
  int n = (int)(i % N);
  int k = (int)(i / N);
  WT[(long)n * K + k] = (__bf16)Win[i];
}

__global__ void __launch_bounds__(256) lengths_kernel(const int* mask, int* lengths) {
  const int S = 2048;
  int b = blockIdx.x;
  __shared__ int red[256];
  int s = 0;
  for (int i = threadIdx.x; i < S; i += 256) s += mask[b * S + i];
  red[threadIdx.x] = s;
  __syncthreads();
  for (int o = 128; o > 0; o >>= 1) {
    if (threadIdx.x < o) red[threadIdx.x] += red[threadIdx.x + o];
    __syncthreads();
  }
  if (threadIdx.x == 0) lengths[b] = red[0];
}

// ---------------------------------------------------------------- q/k GEMM
// One wave computes a 16x64 tile of C = A(MxK) * B(KxN) + bias, A row-major bf16,
// B given pre-transposed (N x K row-major bf16). C stored bf16 row-major.
__global__ void __launch_bounds__(32) gemm_qk_kernel(const __bf16* A, const __bf16* BT,
                                                     const float* bias, __bf16* C,
                                                     int K, int N) {
  int lane = threadIdx.x & 31;
  int m0 = blockIdx.x * 16;
  int n0 = blockIdx.y * 64;
  v8f acc[4];
#pragma unroll
  for (int j = 0; j < 4; ++j) acc[j] = zero_v8f();
  for (int kc = 0; kc < K; kc += 32) {
    v16bf a = ldA(A, K, m0, kc);
#pragma unroll
    for (int j = 0; j < 4; ++j) {
      v16bf b = ldBT(BT, K, n0 + j * 16, kc);
      acc[j] = wmma_bf16(a, b, acc[j]);
    }
  }
  int cl = lane & 15, hi = lane >> 4;
#pragma unroll
  for (int j = 0; j < 4; ++j) {
    int col = n0 + j * 16 + cl;
    float bv = bias[col];
#pragma unroll
    for (int r = 0; r < 8; ++r) {
      int row = m0 + r + 8 * hi;
      C[(long)row * N + col] = (__bf16)(acc[j][r] + bv);
    }
  }
}

// ---------------------------------------------------------------- flash attention
// 8 waves / block, one 16-query tile per block. Wave w computes the score tile for
// key columns [t+16w, t+16w+16) and owns context columns [128w, 128w+128).
__global__ void __launch_bounds__(256) flash_attn_kernel(const __bf16* qbf, const __bf16* kbf,
                                                         const __bf16* hsT, const int* lengths,
                                                         __bf16* ctx) {
  const int S = 2048, D = 512, H = 1024;
  int tid = threadIdx.x, lane = tid & 31, w = tid >> 5;
  int b = blockIdx.y, m0 = blockIdx.x * 16;
  long qrow0 = (long)b * S + m0;
  int len = lengths[b];

  __shared__ float  sS[16 * 128];
  __shared__ __bf16 sP[16 * 128];
  __shared__ float  row_m[16], row_l[16], row_alpha[16];

  if (tid < 16) { row_m[tid] = -__builtin_inff(); row_l[tid] = 0.0f; }

  const __bf16* qbase = qbf + qrow0 * D;
  v16bf qf[16];
#pragma unroll
  for (int kc = 0; kc < 16; ++kc) qf[kc] = ldA(qbase, D, 0, kc * 32);

  v8f acc[8];
#pragma unroll
  for (int nt = 0; nt < 8; ++nt) acc[nt] = zero_v8f();

  const __bf16* kbase = kbf + (long)b * S * D;
  const __bf16* hT = hsT + (long)b * H * S;
  int n0 = w * 128;
  int cl = lane & 15, hi = lane >> 4;
  __syncthreads();

  for (int t = 0; t < S; t += 128) {
    // ---- scores: q(16xD) x k[t+16w .. +16)^T
    v8f sc = zero_v8f();
#pragma unroll
    for (int kc = 0; kc < 16; ++kc) {
      v16bf kb = ldBT(kbase, D, t + 16 * w, kc * 32);
      sc = wmma_bf16(qf[kc], kb, sc);
    }
    int colg = t + 16 * w + cl;
    bool cv = colg < len;
#pragma unroll
    for (int r = 0; r < 8; ++r) {
      float v = cv ? sc[r] * 0.04419417382415922f : -10000.0f;
      sS[(r + 8 * hi) * 128 + 16 * w + cl] = v;
    }
    __syncthreads();
    // ---- online softmax stats (one thread per query row)
    if (tid < 16) {
      float mx = row_m[tid];
      for (int c = 0; c < 128; ++c) mx = fmaxf(mx, sS[tid * 128 + c]);
      float al = __expf(row_m[tid] - mx);
      float s = 0.0f;
      for (int c = 0; c < 128; ++c) {
        float p = __expf(sS[tid * 128 + c] - mx);
        sP[tid * 128 + c] = (__bf16)p;
        s += p;
      }
      row_alpha[tid] = al;
      row_l[tid] = row_l[tid] * al + s;
      row_m[tid] = mx;
    }
    __syncthreads();
    // ---- rescale accumulators, accumulate probs @ hs
    float al[8];
#pragma unroll
    for (int r = 0; r < 8; ++r) al[r] = row_alpha[r + 8 * hi];
#pragma unroll
    for (int nt = 0; nt < 8; ++nt)
#pragma unroll
      for (int r = 0; r < 8; ++r) acc[nt][r] *= al[r];
    v16bf pf[4];
#pragma unroll
    for (int kc = 0; kc < 4; ++kc) pf[kc] = ldA(sP, 128, 0, kc * 32);
#pragma unroll
    for (int nt = 0; nt < 8; ++nt) {
#pragma unroll
      for (int kc = 0; kc < 4; ++kc) {
        v16bf hb = ldBT(hT, S, n0 + nt * 16, t + kc * 32);
        acc[nt] = wmma_bf16(pf[kc], hb, acc[nt]);
      }
    }
    __syncthreads();
  }
  // ---- normalize + store context (bf16)
#pragma unroll
  for (int nt = 0; nt < 8; ++nt) {
#pragma unroll
    for (int r = 0; r < 8; ++r) {
      int row = r + 8 * hi;
      float v = acc[nt][r] / row_l[row];
      ctx[(qrow0 + row) * H + n0 + nt * 16 + cl] = (__bf16)v;
    }
  }
}

// ---------------------------------------------------------------- head: ctx@W1 + LN + relu + W2
__global__ void __launch_bounds__(256) head_kernel(const __bf16* ctx, const __bf16* w1T,
                                                   const float* b1, const float* ln_g,
                                                   const float* ln_b, const float* W2,
                                                   const float* b2, const int* lengths,
                                                   float* token_scores) {
  const int S = 2048, H = 1024;
  int tid = threadIdx.x, lane = tid & 31, w = tid >> 5;
  long row0 = (long)blockIdx.x * 16;
  int b = (int)(row0 / S);
  int len = lengths[b];
  const __bf16* abase = ctx + row0 * H;
  int n0 = w * 64;

  v8f acc[4];
#pragma unroll
  for (int j = 0; j < 4; ++j) acc[j] = zero_v8f();
  for (int kc = 0; kc < H; kc += 32) {
    v16bf a = ldA(abase, H, 0, kc);
#pragma unroll
    for (int j = 0; j < 4; ++j) {
      v16bf bf = ldBT(w1T, H, n0 + j * 16, kc);
      acc[j] = wmma_bf16(a, bf, acc[j]);
    }
  }
  __shared__ float s_sum[16], s_sq[16], s_ts[16];
  if (tid < 16) { s_sum[tid] = 0.0f; s_sq[tid] = 0.0f; s_ts[tid] = 0.0f; }
  __syncthreads();

  int cl = lane & 15, hi = lane >> 4;
  float hv[4][8];
#pragma unroll
  for (int j = 0; j < 4; ++j) {
    int col = n0 + j * 16 + cl;
    float bv = b1[col];
#pragma unroll
    for (int r = 0; r < 8; ++r) hv[j][r] = acc[j][r] + bv;
  }
#pragma unroll
  for (int r = 0; r < 8; ++r) {
    float ps = 0.0f, pq = 0.0f;
#pragma unroll
    for (int j = 0; j < 4; ++j) { ps += hv[j][r]; pq += hv[j][r] * hv[j][r]; }
#pragma unroll
    for (int off = 1; off < 16; off <<= 1) {
      ps += __shfl_xor(ps, off, 32);
      pq += __shfl_xor(pq, off, 32);
    }
    if (cl == 0) { atomicAdd(&s_sum[r + 8 * hi], ps); atomicAdd(&s_sq[r + 8 * hi], pq); }
  }
  __syncthreads();
#pragma unroll
  for (int r = 0; r < 8; ++r) {
    int rowi = r + 8 * hi;
    float mu = s_sum[rowi] * (1.0f / 512.0f);
    float var = s_sq[rowi] * (1.0f / 512.0f) - mu * mu;
    float rstd = rsqrtf(var + 1e-5f);
    float ts = 0.0f;
#pragma unroll
    for (int j = 0; j < 4; ++j) {
      int col = n0 + j * 16 + cl;
      float hn = (hv[j][r] - mu) * rstd * ln_g[col] + ln_b[col];
      hn = fmaxf(hn, 0.0f);
      ts += hn * W2[col];
    }
#pragma unroll
    for (int off = 1; off < 16; off <<= 1) ts += __shfl_xor(ts, off, 32);
    if (cl == 0) atomicAdd(&s_ts[rowi], ts);
  }
  __syncthreads();
  if (tid < 16) {
    long row = row0 + tid;
    int s = (int)(row % S);
    float ts = s_ts[tid] + b2[0];
    token_scores[row] = (s < len) ? ts : -10000.0f;
  }
}

// ---------------------------------------------------------------- rationale span search
__global__ void __launch_bounds__(256) rationale_kernel(const float* token_scores,
                                                        const int* lengths, float* rationale) {
  const int S = 2048;
  int b = blockIdx.x, tid = threadIdx.x;
  __shared__ float cum[2048];
  __shared__ float chunk[256];
  __shared__ float bval[256];
  __shared__ int   bidx[256];

  float loc[8];
  float run = 0.0f;
#pragma unroll
  for (int j = 0; j < 8; ++j) { run += token_scores[b * S + tid * 8 + j]; loc[j] = run; }
  chunk[tid] = run;
  __syncthreads();
  float x = run;
  for (int o = 1; o < 256; o <<= 1) {
    float y = (tid >= o) ? chunk[tid - o] : 0.0f;
    __syncthreads();
    x += y;
    chunk[tid] = x;
    __syncthreads();
  }
  float off = x - run;
#pragma unroll
  for (int j = 0; j < 8; ++j) cum[tid * 8 + j] = loc[j] + off;
  __syncthreads();

  int vl = lengths[b];
  float best = -__builtin_inff();
  int bi = 0x40000000;
  for (int idx = tid; idx < 18 * S; idx += 256) {
    int Li = idx / S, s = idx - Li * S;
    int L = 3 + Li;
    int end = s + L - 1;
    if (end > S - 1) end = S - 1;
    bool valid = (L <= vl) && (s <= vl - L);
    if (valid) {
      float v = cum[end] / (float)L + 0.01f * ((float)L / 20.0f);
      if (v > best || (v == best && idx < bi)) { best = v; bi = idx; }
    }
  }
  bval[tid] = best;
  bidx[tid] = bi;
  __syncthreads();
  for (int o = 128; o > 0; o >>= 1) {
    if (tid < o) {
      float v2 = bval[tid + o];
      int i2 = bidx[tid + o];
      if (v2 > bval[tid] || (v2 == bval[tid] && i2 < bidx[tid])) { bval[tid] = v2; bidx[tid] = i2; }
    }
    __syncthreads();
  }
  int bestIdx = bidx[0];
  int bestL = 3 + bestIdx / S;
  int bestS = bestIdx % S;
  bool shortB = (vl <= 3);
  for (int p = tid; p < S; p += 256) {
    bool in = shortB ? (p < vl) : (p >= bestS && p < bestS + bestL);
    rationale[b * S + p] = in ? 1.0f : 0.0f;
  }
}

// ---------------------------------------------------------------- token_probs softmax
__global__ void __launch_bounds__(256) softmax_row_kernel(const float* ts, float* probs) {
  const int S = 2048;
  int b = blockIdx.x, tid = threadIdx.x;
  __shared__ float red[256];
  float mx = -__builtin_inff();
  for (int i = tid; i < S; i += 256) mx = fmaxf(mx, ts[b * S + i]);
  red[tid] = mx;
  __syncthreads();
  for (int o = 128; o > 0; o >>= 1) { if (tid < o) red[tid] = fmaxf(red[tid], red[tid + o]); __syncthreads(); }
  mx = red[0];
  __syncthreads();
  float s = 0.0f;
  for (int i = tid; i < S; i += 256) s += __expf(ts[b * S + i] - mx);
  red[tid] = s;
  __syncthreads();
  for (int o = 128; o > 0; o >>= 1) { if (tid < o) red[tid] += red[tid + o]; __syncthreads(); }
  float inv = 1.0f / red[0];
  for (int i = tid; i < S; i += 256) probs[b * S + i] = __expf(ts[b * S + i] - mx) * inv;
}

// ---------------------------------------------------------------- attended + pooled
__global__ void __launch_bounds__(256) pool_kernel(const float* hs, const float* rationale,
                                                   float* attended, float* pooled) {
  const int S = 2048, H = 1024;
  int b = blockIdx.x, tid = threadIdx.x;
  __shared__ float red[256];
  float c = 0.0f;
  for (int i = tid; i < S; i += 256) c += rationale[b * S + i];
  red[tid] = c;
  __syncthreads();
  for (int o = 128; o > 0; o >>= 1) { if (tid < o) red[tid] += red[tid + o]; __syncthreads(); }
  float rlen = red[0] + 1e-6f;
  int h0 = tid * 4;
  float a0 = 0.0f, a1 = 0.0f, a2 = 0.0f, a3 = 0.0f;
  const float* base = hs + (long)b * S * H;
  float* abase = attended + (long)b * S * H;
  for (int s = 0; s < S; ++s) {
    float r = rationale[b * S + s];
    float4 v = *(const float4*)(base + (long)s * H + h0);
    float4 a;
    a.x = v.x * r; a.y = v.y * r; a.z = v.z * r; a.w = v.w * r;
    *(float4*)(abase + (long)s * H + h0) = a;
    a0 += a.x; a1 += a.y; a2 += a.z; a3 += a.w;
  }
  pooled[b * H + h0 + 0] = a0 / rlen;
  pooled[b * H + h0 + 1] = a1 / rlen;
  pooled[b * H + h0 + 2] = a2 / rlen;
  pooled[b * H + h0 + 3] = a3 / rlen;
}

// ---------------------------------------------------------------- launch
extern "C" void kernel_launch(void* const* d_in, const int* in_sizes, int n_in,
                              void* d_out, int out_size, void* d_ws, size_t ws_size,
                              hipStream_t stream) {
  const int B = 16, S = 2048, H = 1024, D = 512;
  const float* hs   = (const float*)d_in[0];
  const int*   mask = (const int*)d_in[1];
  const float* Wq   = (const float*)d_in[2];
  const float* bq   = (const float*)d_in[3];
  const float* Wk   = (const float*)d_in[4];
  const float* bk   = (const float*)d_in[5];
  const float* W1   = (const float*)d_in[6];
  const float* b1   = (const float*)d_in[7];
  const float* ln_g = (const float*)d_in[8];
  const float* ln_b = (const float*)d_in[9];
  const float* W2   = (const float*)d_in[10];
  const float* b2   = (const float*)d_in[11];

  char* ws = (char*)d_ws;
  size_t off = 0;
  auto alloc = [&](size_t bytes) { size_t o = off; off += (bytes + 255) & ~(size_t)255; return o; };
  __bf16* hsb  = (__bf16*)(ws + alloc((size_t)B * S * H * 2));
  __bf16* hsT  = (__bf16*)(ws + alloc((size_t)B * H * S * 2));
  __bf16* qbf  = (__bf16*)(ws + alloc((size_t)B * S * D * 2));
  __bf16* kbf  = (__bf16*)(ws + alloc((size_t)B * S * D * 2));
  __bf16* ctx  = (__bf16*)(ws + alloc((size_t)B * S * H * 2));
  __bf16* wqT  = (__bf16*)(ws + alloc((size_t)D * H * 2));
  __bf16* wkT  = (__bf16*)(ws + alloc((size_t)D * H * 2));
  __bf16* w1T  = (__bf16*)(ws + alloc((size_t)D * H * 2));
  int*    lens = (int*)(ws + alloc(64));

  float* out_ts     = (float*)d_out;
  float* out_rat    = out_ts + (size_t)B * S;
  float* out_probs  = out_rat + (size_t)B * S;
  float* out_att    = out_probs + (size_t)B * S;
  float* out_pooled = out_att + (size_t)B * S * H;

  lengths_kernel<<<B, 256, 0, stream>>>(mask, lens);
  {
    long n = (long)B * S * H;
    convert_hs_kernel<<<(unsigned)((n + 255) / 256), 256, 0, stream>>>(hs, hsb, hsT);
  }
  transposeW_kernel<<<(H * D + 255) / 256, 256, 0, stream>>>(Wq, wqT, H, D);
  transposeW_kernel<<<(H * D + 255) / 256, 256, 0, stream>>>(Wk, wkT, H, D);
  transposeW_kernel<<<(H * D + 255) / 256, 256, 0, stream>>>(W1, w1T, H, D);

  dim3 gq(B * S / 16, D / 64);
  gemm_qk_kernel<<<gq, 32, 0, stream>>>(hsb, wqT, bq, qbf, H, D);
  gemm_qk_kernel<<<gq, 32, 0, stream>>>(hsb, wkT, bk, kbf, H, D);

  flash_attn_kernel<<<dim3(S / 16, B), 256, 0, stream>>>(qbf, kbf, hsT, lens, ctx);

  head_kernel<<<B * S / 16, 256, 0, stream>>>(ctx, w1T, b1, ln_g, ln_b, W2, b2, lens, out_ts);

  rationale_kernel<<<B, 256, 0, stream>>>(out_ts, lens, out_rat);
  softmax_row_kernel<<<B, 256, 0, stream>>>(out_ts, out_probs);
  pool_kernel<<<B, 256, 0, stream>>>(hs, out_rat, out_att, out_pooled);

  (void)in_sizes; (void)n_in; (void)out_size; (void)ws_size;
}
